// GP_1039382086202
// MI455X (gfx1250) — matrix operand order repaired and checked by reading, not compiled, not run on hardware
//
#include <hip/hip_runtime.h>
#include <hip/hip_bf16.h>
#include <math.h>

#define B_    4
#define C_    512
#define H_    40
#define W_    40
#define N_    1600          // H_*W_
#define GP_   64
#define NB_   64            // inversion panel width
#define NBLK_ (N_/NB_)      // 25
#define SIGMA_ 0.1f
#define EPS_   1e-6f
#define NN_   ((size_t)N_*(size_t)N_)

typedef __attribute__((ext_vector_type(2)))  float    v2f;
typedef __attribute__((ext_vector_type(8)))  float    v8f;
typedef __attribute__((ext_vector_type(8)))  _Float16 v8h;
typedef __attribute__((ext_vector_type(16))) _Float16 v16h;

// ---------------- WMMA wrappers (CDNA5 / gfx1250, wave32) ----------------
__device__ __forceinline__ v8f wmma_f32_k4(v2f a, v2f b, v8f c) {
  return __builtin_amdgcn_wmma_f32_16x16x4_f32(false, a, false, b, (short)0, c,
                                               false, false);
}
__device__ __forceinline__ v8f wmma_f16_k32(v16h a, v16h b, v8f c) {
  return __builtin_amdgcn_wmma_f32_16x16x32_f16(false, a, false, b, (short)0, c,
                                                false, false);
}

// Async global->LDS copy of one 64x32 f32 tile (8KB), one wave (32 lanes).
// src row kk (0..63) is srcRow0 + kk*srcStride floats; LDS layout [64][32].
__device__ __forceinline__ void async_stage_64x32(const float* srcRow0, size_t srcStride,
                                                  float* ldsTile, int lane) {
  unsigned ldsbase = (unsigned)(size_t)(void*)ldsTile;  // low 32 bits = LDS offset
#pragma unroll
  for (int it = 0; it < 16; ++it) {
    int flat = it*128 + lane*4;           // float index within 64x32 tile
    int kk   = flat >> 5;
    int col  = flat & 31;
    const float* ga = srcRow0 + (size_t)kk*srcStride + col;
    asm volatile("global_load_async_to_lds_b128 %0, %1, off"
                 :: "v"(ldsbase + (unsigned)(flat*4)), "v"(ga) : "memory");
  }
}
__device__ __forceinline__ void wait_async0() {
  asm volatile("s_wait_asynccnt 0" ::: "memory");
}
__device__ __forceinline__ void wait_ds0() {
  asm volatile("s_wait_dscnt 0" ::: "memory");
}

// f16 A fragment (16x32 tile, row-major source, ld = C_)
__device__ __forceinline__ v16h load_afrag16(const _Float16* base, int k0, int kh) {
  v8h lo = *(const v8h*)(base + k0 + 8*kh);
  v8h hi = *(const v8h*)(base + k0 + 16 + 8*kh);
  v16h a;
#pragma unroll
  for (int i = 0; i < 8; ++i) { a[i] = lo[i]; a[i+8] = hi[i]; }
  return a;
}
// f16 B fragment (32x16 tile, column data contiguous in source rows, ld = C_)
__device__ __forceinline__ v16h load_bfrag16(const _Float16* base, int k0, int kh) {
  v8h lo = *(const v8h*)(base + k0 + 8*kh);
  v8h hi = *(const v8h*)(base + k0 + 16 + 8*kh);
  v16h b;
#pragma unroll
  for (int i = 0; i < 8; ++i) { b[i] = lo[i]; b[i+8] = hi[i]; }
  return b;
}

// ---------------- norms: |x_p|, |y_p| per point ----------------
__global__ void norms_kernel(const float* __restrict__ x, const float* __restrict__ y,
                             float* __restrict__ normx, float* __restrict__ normy) {
  __shared__ float sx[256], sy[256];
  int p  = blockIdx.x % N_;
  int bb = blockIdx.x / N_;
  int tid = threadIdx.x;
  float ax = 0.f, ay = 0.f;
  for (int ch = tid; ch < C_; ch += 256) {
    size_t i = ((size_t)bb*C_ + ch)*N_ + p;
    float vx = x[i]; ax += vx*vx;
    float vy = y[i]; ay += vy*vy;
  }
  sx[tid] = ax; sy[tid] = ay; __syncthreads();
  for (int s = 128; s > 0; s >>= 1) {
    if (tid < s) { sx[tid] += sx[tid+s]; sy[tid] += sy[tid+s]; }
    __syncthreads();
  }
  if (tid == 0) {
    normx[bb*N_ + p] = sqrtf(sx[0]);
    normy[bb*N_ + p] = sqrtf(sy[0]);
  }
}

// ---------------- transpose (b,c,p) -> (b,p,c) normalized f16 ----------------
__global__ void pack_kernel(const float* __restrict__ x, const float* __restrict__ y,
                            const float* __restrict__ normx, const float* __restrict__ normy,
                            _Float16* __restrict__ xnh, _Float16* __restrict__ ynh) {
  int idx = blockIdx.x*blockDim.x + threadIdx.x;
  if (idx >= B_*C_*N_) return;
  int p  = idx % N_;
  int ch = (idx / N_) % C_;
  int bb = idx / (N_*C_);
  float rx = 1.0f / normx[bb*N_ + p];
  float ry = 1.0f / normy[bb*N_ + p];
  xnh[((size_t)bb*N_ + p)*C_ + ch] = (_Float16)(x[idx]*rx);
  ynh[((size_t)bb*N_ + p)*C_ + ch] = (_Float16)(y[idx]*ry);
}

// ---------------- positional encoding fr[p][d] (shared by all batches) --------
__global__ void pos_enc_kernel(const float* __restrict__ pos_w, const float* __restrict__ pos_b,
                               float* __restrict__ fr) {
  int p = blockIdx.x, d = threadIdx.x;
  int i = p / W_, j = p % W_;
  float gy = -1.0f + 1.0f/H_ + i * ((2.0f - 2.0f/H_) / (H_ - 1));
  float gx = -1.0f + 1.0f/W_ + j * ((2.0f - 2.0f/W_) / (W_ - 1));
  float t = pos_w[d*2+0]*gx + pos_w[d*2+1]*gy + pos_b[d];
  fr[p*GP_ + d] = cosf(25.132741228718345f * t);   // cos(8*pi*t)
}

// ---------------- Gram: K = exp(c - 1), c = dot*(na*nb)/(na*nb+eps) -----------
__device__ __forceinline__ void gram_store(v8f acc, int bb, int rbase, int cbase,
                                           const float* nA, const float* nB,
                                           float* Kout, int addNoise, int m, int kh) {
  int gcol = cbase + m;
  float nb = nB[bb*N_ + gcol];
#pragma unroll
  for (int r = 0; r < 8; ++r) {
    int grow = rbase + r + 8*kh;
    float na = nA[bb*N_ + grow];
    float s  = na*nb;
    float cc = acc[r] * (s/(s+EPS_));
    float val = __expf(cc - 1.0f);
    if (addNoise && grow == gcol) val += SIGMA_;
    Kout[(size_t)bb*NN_ + (size_t)grow*N_ + gcol] = val;
  }
}

__global__ void gram_kernel(const _Float16* __restrict__ Ah, const _Float16* __restrict__ Bh,
                            const float* __restrict__ nA, const float* __restrict__ nB,
                            float* __restrict__ Kout, int addNoise) {
  int lane = threadIdx.x & 31;
  int tc2 = blockIdx.x, tr2 = blockIdx.y, bb = blockIdx.z;
  int m  = lane & 15;
  int kh = lane >> 4;
  const _Float16* A0 = Ah + ((size_t)bb*N_ + tr2*32 + m)*C_;
  const _Float16* A1 = A0 + (size_t)16*C_;
  const _Float16* B0 = Bh + ((size_t)bb*N_ + tc2*32 + m)*C_;
  const _Float16* B1 = B0 + (size_t)16*C_;
  v8f c00 = {}, c01 = {}, c10 = {}, c11 = {};
  for (int k0 = 0; k0 < C_; k0 += 32) {
    v16h a0 = load_afrag16(A0, k0, kh);
    v16h a1 = load_afrag16(A1, k0, kh);
    v16h b0 = load_bfrag16(B0, k0, kh);
    v16h b1 = load_bfrag16(B1, k0, kh);
    c00 = wmma_f16_k32(a0, b0, c00);
    c01 = wmma_f16_k32(a0, b1, c01);
    c10 = wmma_f16_k32(a1, b0, c10);
    c11 = wmma_f16_k32(a1, b1, c11);
  }
  gram_store(c00, bb, tr2*32,      tc2*32,      nA, nB, Kout, addNoise, m, kh);
  gram_store(c01, bb, tr2*32,      tc2*32 + 16, nA, nB, Kout, addNoise, m, kh);
  gram_store(c10, bb, tr2*32 + 16, tc2*32,      nA, nB, Kout, addNoise, m, kh);
  gram_store(c11, bb, tr2*32 + 16, tc2*32 + 16, nA, nB, Kout, addNoise, m, kh);
}

// ---------------- 64x64 diagonal block inverse (Gauss-Jordan in LDS) ----------
__global__ void diag_inv_kernel(const float* __restrict__ A, float* __restrict__ dinv, int kb) {
  __shared__ float s[64][65];
  int j = threadIdx.x;
  int bb = blockIdx.x;
  const float* Ab = A + (size_t)bb*NN_;
  for (int i = 0; i < 64; ++i)
    s[i][j] = Ab[(size_t)(kb*NB_ + i)*N_ + kb*NB_ + j];
  __syncthreads();
  for (int k = 0; k < 64; ++k) {
    float pivot = 1.0f / s[k][k];
    float rk = s[k][j];
    __syncthreads();
    if (j != k) {
      for (int i = 0; i < 64; ++i) if (i != k) s[i][j] -= s[i][k]*pivot*rk;
      s[k][j] = -pivot*rk;
    }
    __syncthreads();
    if (j == k) {
      for (int i = 0; i < 64; ++i) if (i != k) s[i][k] *= pivot;
      s[k][k] = pivot;
    }
    __syncthreads();
  }
  for (int i = 0; i < 64; ++i)
    dinv[((size_t)bb*NB_ + i)*NB_ + j] = s[i][j];
}

// ---------------- save old row panel A[k,:] ----------------
__global__ void copy_rowpanel_kernel(const float* __restrict__ A, float* __restrict__ temp, int kb) {
  int idx = blockIdx.x*blockDim.x + threadIdx.x;
  if (idx >= B_*NB_*N_) return;
  int col = idx % N_;
  int row = (idx / N_) % NB_;
  int bb  = idx / (N_*NB_);
  temp[idx] = A[(size_t)bb*NN_ + (size_t)(kb*NB_ + row)*N_ + col];
}

// ---------------- column panel: A[i,k] <- A[i,k] @ Dinv (in place) ------------
__global__ void col_update_kernel(float* __restrict__ A, const float* __restrict__ dinv, int kb) {
  int rs = blockIdx.x, bb = blockIdx.z;
  if ((rs >> 2) == kb) return;          // skip diagonal block row (uniform)
  int lane = threadIdx.x & 31;
  int m = lane & 15, kh = lane >> 4;
  float* Ab = A + (size_t)bb*NN_;
  const float* Db = dinv + (size_t)bb*NB_*NB_;
  const float* Arow = Ab + (size_t)(rs*16 + m)*N_ + kb*NB_;
  v2f areg[16];
#pragma unroll
  for (int t = 0; t < 16; ++t) { int kk = 4*t + 2*kh; areg[t][0] = Arow[kk]; areg[t][1] = Arow[kk+1]; }
  v8f acc[4];
#pragma unroll
  for (int cb = 0; cb < 4; ++cb) {
    v8f c = {};
#pragma unroll
    for (int t = 0; t < 16; ++t) {
      int kk = 4*t + 2*kh;
      v2f bf; bf[0] = Db[kk*NB_ + cb*16 + m]; bf[1] = Db[(kk+1)*NB_ + cb*16 + m];
      c = wmma_f32_k4(areg[t], bf, c);
    }
    acc[cb] = c;
  }
#pragma unroll
  for (int cb = 0; cb < 4; ++cb)
#pragma unroll
    for (int r = 0; r < 8; ++r)
      Ab[(size_t)(rs*16 + r + 8*kh)*N_ + kb*NB_ + cb*16 + m] = acc[cb][r];
}

// ---------------- Schur: A[i,j] -= A[i,k]_new @ A[k,j]_old --------------------
// 2x2 register-blocked; B-panel (64x32 of temp) staged into LDS asynchronously.
__global__ void schur_kernel(float* __restrict__ A, const float* __restrict__ temp, int kb) {
  __shared__ float bpanel[64*32];
  int tc2 = blockIdx.x, tr2 = blockIdx.y, bb = blockIdx.z;
  if ((tr2 >> 1) == kb || (tc2 >> 1) == kb) return;   // 32 | 64 -> block-uniform
  int lane = threadIdx.x & 31;
  int m = lane & 15, kh = lane >> 4;
  float* Ab = A + (size_t)bb*NN_;
  const float* Tb = temp + (size_t)bb*NB_*N_;

  // kick off async B-panel copy: temp rows 0..63, cols tc2*32..+32 -> LDS
  async_stage_64x32(Tb + tc2*32, N_, bpanel, lane);

  // overlap: load A fragments (negated) and C tiles from global
  const float* Arow0 = Ab + (size_t)(tr2*32 + m)*N_ + kb*NB_;
  const float* Arow1 = Arow0 + (size_t)16*N_;
  v2f a0[16], a1[16];
#pragma unroll
  for (int t = 0; t < 16; ++t) {       // negate A frags (f32 WMMA has no NEG on A/B)
    int kk = 4*t + 2*kh;
    a0[t][0] = -Arow0[kk]; a0[t][1] = -Arow0[kk+1];
    a1[t][0] = -Arow1[kk]; a1[t][1] = -Arow1[kk+1];
  }
  v8f c00, c01, c10, c11;
#pragma unroll
  for (int r = 0; r < 8; ++r) {
    size_t r0 = (size_t)(tr2*32 + r + 8*kh)*N_;
    size_t r1 = r0 + (size_t)16*N_;
    c00[r] = Ab[r0 + tc2*32 + m];      c01[r] = Ab[r0 + tc2*32 + 16 + m];
    c10[r] = Ab[r1 + tc2*32 + m];      c11[r] = Ab[r1 + tc2*32 + 16 + m];
  }
  wait_async0();                       // B-panel resident in LDS
#pragma unroll
  for (int t = 0; t < 16; ++t) {
    int kk = 4*t + 2*kh;
    v2f b0; b0[0] = bpanel[kk*32 + m];      b0[1] = bpanel[(kk+1)*32 + m];
    v2f b1; b1[0] = bpanel[kk*32 + 16 + m]; b1[1] = bpanel[(kk+1)*32 + 16 + m];
    c00 = wmma_f32_k4(a0[t], b0, c00);
    c01 = wmma_f32_k4(a0[t], b1, c01);
    c10 = wmma_f32_k4(a1[t], b0, c10);
    c11 = wmma_f32_k4(a1[t], b1, c11);
  }
#pragma unroll
  for (int r = 0; r < 8; ++r) {
    size_t r0 = (size_t)(tr2*32 + r + 8*kh)*N_;
    size_t r1 = r0 + (size_t)16*N_;
    Ab[r0 + tc2*32 + m]      = c00[r];  Ab[r0 + tc2*32 + 16 + m] = c01[r];
    Ab[r1 + tc2*32 + m]      = c10[r];  Ab[r1 + tc2*32 + 16 + m] = c11[r];
  }
}

// ---------------- row panel: A[k,j] <- -Dinv @ A[k,j]_old ; A[k,k] <- Dinv ----
__global__ void row_update_kernel(float* __restrict__ A, const float* __restrict__ temp,
                                  const float* __restrict__ dinv, int kb) {
  int ct = blockIdx.x, rt = blockIdx.y, bb = blockIdx.z;
  int lane = threadIdx.x & 31;
  int m = lane & 15, kh = lane >> 4;
  float* Ab = A + (size_t)bb*NN_;
  const float* Db = dinv + (size_t)bb*NB_*NB_;
  if ((ct >> 2) == kb) {                      // diagonal block: write Dinv
    int cl = (ct - kb*4)*16 + m;
#pragma unroll
    for (int r = 0; r < 8; ++r)
      Ab[(size_t)(kb*NB_ + rt*16 + r + 8*kh)*N_ + ct*16 + m] =
          Db[(rt*16 + r + 8*kh)*NB_ + cl];
    return;
  }
  const float* Tb = temp + (size_t)bb*NB_*N_;
  const float* Drow = Db + (rt*16 + m)*NB_;
  v2f areg[16];
#pragma unroll
  for (int t = 0; t < 16; ++t) { int kk = 4*t + 2*kh; areg[t][0] = -Drow[kk]; areg[t][1] = -Drow[kk+1]; }
  v8f c = {};
#pragma unroll
  for (int t = 0; t < 16; ++t) {
    int kk = 4*t + 2*kh;
    v2f bf; bf[0] = Tb[(size_t)kk*N_ + ct*16 + m]; bf[1] = Tb[(size_t)(kk+1)*N_ + ct*16 + m];
    c = wmma_f32_k4(areg[t], bf, c);
  }
#pragma unroll
  for (int r = 0; r < 8; ++r)
    Ab[(size_t)(kb*NB_ + rt*16 + r + 8*kh)*N_ + ct*16 + m] = c[r];
}

// ---------------- M = K_xy @ K_yy_inv (2x2 blocked, async-LDS B, k-chunked) ---
__global__ void gemm_M_kernel(const float* __restrict__ Kxy, const float* __restrict__ Ainv,
                              float* __restrict__ M) {
  __shared__ float bpanel[64*32];
  int tc2 = blockIdx.x, tr2 = blockIdx.y, bb = blockIdx.z;
  int lane = threadIdx.x & 31;
  int m = lane & 15, kh = lane >> 4;
  const float* Arow0 = Kxy + (size_t)bb*NN_ + (size_t)(tr2*32 + m)*N_;
  const float* Arow1 = Arow0 + (size_t)16*N_;
  const float* Bm    = Ainv + (size_t)bb*NN_;
  v8f c00 = {}, c01 = {}, c10 = {}, c11 = {};
  for (int chunk = 0; chunk < N_/64; ++chunk) {
    wait_ds0();                                 // prior chunk's LDS reads done
    async_stage_64x32(Bm + (size_t)(chunk*64)*N_ + tc2*32, N_, bpanel, lane);
    if (chunk + 1 < N_/64) {                    // prefetch next A panel
      __builtin_prefetch(Arow0 + (chunk+1)*64, 0, 0);
      __builtin_prefetch(Arow1 + (chunk+1)*64, 0, 0);
    }
    wait_async0();                              // B chunk resident in LDS
#pragma unroll
    for (int t = 0; t < 16; ++t) {
      int kk4 = 4*t + 2*kh;                     // k within chunk
      int kk  = chunk*64 + kk4;
      v2f A0; A0[0] = Arow0[kk]; A0[1] = Arow0[kk+1];
      v2f A1; A1[0] = Arow1[kk]; A1[1] = Arow1[kk+1];
      v2f b0; b0[0] = bpanel[kk4*32 + m];      b0[1] = bpanel[(kk4+1)*32 + m];
      v2f b1; b1[0] = bpanel[kk4*32 + 16 + m]; b1[1] = bpanel[(kk4+1)*32 + 16 + m];
      c00 = wmma_f32_k4(A0, b0, c00);
      c01 = wmma_f32_k4(A0, b1, c01);
      c10 = wmma_f32_k4(A1, b0, c10);
      c11 = wmma_f32_k4(A1, b1, c11);
    }
  }
  float* Mb = M + (size_t)bb*NN_;
#pragma unroll
  for (int r = 0; r < 8; ++r) {
    size_t r0 = (size_t)(tr2*32 + r + 8*kh)*N_;
    size_t r1 = r0 + (size_t)16*N_;
    Mb[r0 + tc2*32 + m]      = c00[r];  Mb[r0 + tc2*32 + 16 + m] = c01[r];
    Mb[r1 + tc2*32 + m]      = c10[r];  Mb[r1 + tc2*32 + 16 + m] = c11[r];
  }
}

// ---------------- mu = M @ fr -> out channels [0,64) ----------------
__global__ void mu_kernel(const float* __restrict__ M, const float* __restrict__ fr,
                          float* __restrict__ out) {
  int tcol = blockIdx.x, trow = blockIdx.y, bb = blockIdx.z;
  int lane = threadIdx.x & 31;
  int m = lane & 15, kh = lane >> 4;
  const float* Arow = M + (size_t)bb*NN_ + (size_t)(trow*16 + m)*N_;
  v8f acc = {};
  for (int k0 = 0; k0 < N_; k0 += 4) {
    int kk = k0 + 2*kh;
    v2f a; a[0] = Arow[kk]; a[1] = Arow[kk+1];
    v2f b; b[0] = fr[kk*GP_ + tcol*16 + m]; b[1] = fr[(kk+1)*GP_ + tcol*16 + m];
    acc = wmma_f32_k4(a, b, acc);
  }
  int d = tcol*16 + m;
#pragma unroll
  for (int r = 0; r < 8; ++r) {
    int p = trow*16 + r + 8*kh;
    out[((size_t)bb*89 + d)*N_ + p] = acc[r];
  }
}

// ---------------- local 5x5 cov -> channels [64,89) ----------------
__global__ void cov_local_kernel(const _Float16* __restrict__ xnh, const float* __restrict__ normx,
                                 const float* __restrict__ M, const float* __restrict__ Kxy,
                                 float* __restrict__ out) {
  __shared__ float redA[256], redB[256];
  int p = blockIdx.x, bb = blockIdx.y;
  int tid = threadIdx.x;
  int pi = p / W_, pj = p % W_;
  const _Float16* xp = xnh + ((size_t)bb*N_ + p)*C_;
  float nap = normx[bb*N_ + p];
  const float* Mrow = M + (size_t)bb*NN_ + (size_t)p*N_;
  for (int kk = 0; kk < 25; ++kk) {
    int qi = pi + kk/5 - 2;
    int qj = pj + kk%5 - 2;
    float* op = out + ((size_t)bb*89 + 64 + kk)*N_ + p;
    if (qi >= 0 && qi < H_ && qj >= 0 && qj < W_) {
      int q = qi*W_ + qj;
      const _Float16* xq = xnh + ((size_t)bb*N_ + q)*C_;
      float dot = 0.f;
      for (int c2 = tid; c2 < C_; c2 += 256) dot += (float)xp[c2]*(float)xq[c2];
      float mk = 0.f;
      const float* Kq = Kxy + (size_t)bb*NN_ + (size_t)q*N_;
      for (int t = tid; t < N_; t += 256) mk += Mrow[t]*Kq[t];
      redA[tid] = dot; redB[tid] = mk; __syncthreads();
      for (int s = 128; s > 0; s >>= 1) {
        if (tid < s) { redA[tid] += redA[tid+s]; redB[tid] += redB[tid+s]; }
        __syncthreads();
      }
      if (tid == 0) {
        float s2 = nap * normx[bb*N_ + q];
        float cs = redA[0] * (s2/(s2+EPS_));
        *op = __expf(cs - 1.0f) - redB[0];
      }
      __syncthreads();
    } else {
      if (tid == 0) *op = 0.f;
      __syncthreads();
    }
  }
}

// ---------------- host launch ----------------
extern "C" void kernel_launch(void* const* d_in, const int* in_sizes, int n_in,
                              void* d_out, int out_size, void* d_ws, size_t ws_size,
                              hipStream_t stream) {
  (void)in_sizes; (void)n_in; (void)out_size; (void)ws_size;
  const float* x     = (const float*)d_in[0];
  const float* y     = (const float*)d_in[1];
  const float* pos_w = (const float*)d_in[2];
  const float* pos_b = (const float*)d_in[3];
  float* out = (float*)d_out;

  char* ws = (char*)d_ws;
  size_t off = 0;
  auto take = [&](size_t bytes) -> void* {
    void* p = ws + off;
    off += (bytes + 255) & ~(size_t)255;
    return p;
  };
  float*    normx = (float*)take((size_t)B_*N_*4);
  float*    normy = (float*)take((size_t)B_*N_*4);
  _Float16* xnh   = (_Float16*)take((size_t)B_*N_*C_*2);
  _Float16* ynh   = (_Float16*)take((size_t)B_*N_*C_*2);
  float*    fr    = (float*)take((size_t)N_*GP_*4);
  float*    Kxy   = (float*)take((size_t)B_*NN_*4);
  float*    Ayy   = (float*)take((size_t)B_*NN_*4);
  float*    Mm    = (float*)take((size_t)B_*NN_*4);
  float*    temp  = (float*)take((size_t)B_*NB_*N_*4);
  float*    dinv  = (float*)take((size_t)B_*NB_*NB_*4);

  norms_kernel<<<B_*N_, 256, 0, stream>>>(x, y, normx, normy);
  {
    int total = B_*C_*N_;
    pack_kernel<<<(total+255)/256, 256, 0, stream>>>(x, y, normx, normy, xnh, ynh);
  }
  pos_enc_kernel<<<N_, GP_, 0, stream>>>(pos_w, pos_b, fr);

  dim3 gtile(N_/32, N_/32, B_);
  gram_kernel<<<gtile, 32, 0, stream>>>(ynh, ynh, normy, normy, Ayy, 1); // K_yy + sigma*I
  gram_kernel<<<gtile, 32, 0, stream>>>(xnh, ynh, normx, normy, Kxy, 0); // K_xy

  for (int kb = 0; kb < NBLK_; ++kb) {
    diag_inv_kernel<<<B_, 64, 0, stream>>>(Ayy, dinv, kb);
    {
      int total = B_*NB_*N_;
      copy_rowpanel_kernel<<<(total+255)/256, 256, 0, stream>>>(Ayy, temp, kb);
    }
    col_update_kernel<<<dim3(N_/16, 1, B_), 32, 0, stream>>>(Ayy, dinv, kb);
    schur_kernel<<<dim3(N_/32, N_/32, B_), 32, 0, stream>>>(Ayy, temp, kb);
    row_update_kernel<<<dim3(N_/16, 4, B_), 32, 0, stream>>>(Ayy, temp, dinv, kb);
  }

  gemm_M_kernel<<<dim3(N_/32, N_/32, B_), 32, 0, stream>>>(Kxy, Ayy, Mm);
  mu_kernel<<<dim3(GP_/16, N_/16, B_), 32, 0, stream>>>(Mm, fr, out);
  cov_local_kernel<<<dim3(N_, B_), 256, 0, stream>>>(xnh, normx, Mm, Kxy, out);
}
